// LearnableHash_87007447482483
// MI455X (gfx1250) — matrix operand order repaired and checked by reading, not compiled, not run on hardware
//
#include <hip/hip_runtime.h>
#include <hip/hip_bf16.h>
#include <math.h>

typedef _Float16 half_t;
typedef _Float16 v16h __attribute__((ext_vector_type(16)));
typedef _Float16 v8h  __attribute__((ext_vector_type(8)));
typedef float    v8f  __attribute__((ext_vector_type(8)));
typedef float    v4f  __attribute__((ext_vector_type(4)));

#define RES     160
#define NF      64
#define FDIM    32
#define RADIUSF 1.3f
#define NI      256
#define NRAYS   1024
#define NPTS    (NRAYS * NI)
#define STEPF   (2.0f * RADIUSF / (float)NI)

// ---------------- WMMA helpers ----------------

__device__ __forceinline__ v8f wmma16(v16h a, v16h b, v8f c) {
    // D = A(16x32 f16) * B(32x16 f16) + C(16x16 f32)
    return __builtin_amdgcn_wmma_f32_16x16x32_f16(
        /*neg_a=*/false, a, /*neg_b=*/false, b,
        /*c_mod=*/(short)0, c, /*reuse_a=*/false, /*reuse_b=*/false);
}

// A-operand layout (ISA 7.12.2, 16-bit A 16x32): lane l holds row M=l&15,
// halves h<8 -> K = off8 + h, halves h>=8 -> K = off8 + 8 + h, off8 = (l<16?0:8).
// So a lane's 16 halves are two contiguous 16-byte chunks at K=off8 and K=off8+16.
__device__ __forceinline__ v16h load_a16(const half_t* p0, const half_t* p1) {
    v8h lo = *(const v8h*)p0;
    v8h hi = *(const v8h*)p1;
    v16h r;
#pragma unroll
    for (int i = 0; i < 8; ++i) { r[i] = lo[i]; r[8 + i] = hi[i]; }
    return r;
}

__device__ __forceinline__ int clampi(int v, int lo, int hi) {
    return v < lo ? lo : (v > hi ? hi : v);
}

// ---------------- Kernel 0: pack weights into WMMA B-operand layout ----------------
// B layout (16-bit B 32x16): lane l, half h -> K = ktile*32 + (l<16?0:8) + h + (h<8?0:8), N = ntile*16 + (l&15).
// Packed lane-major: entry ((kt*NT+nt)*32 + lane) holds 16 contiguous halves (32B) -> coalesced b128 loads.
// Half offsets: Ws1@0 (1x4 tiles), Ws2@2048 (2x1), Wc1@3072 (1x4), Wc2@5120 (2x4), Wc3@9216 (2x1).

__global__ void __launch_bounds__(256) prepack_weights(
    const float* __restrict__ Ws1, const float* __restrict__ Ws2,
    const float* __restrict__ Wc1, const float* __restrict__ Wc2,
    const float* __restrict__ Wc3, half_t* __restrict__ packed) {
    int g = blockIdx.x * blockDim.x + threadIdx.x;
    if (g >= 640) return;
    const float* W; int Krows, Ncols, NT, off, base;
    if (g < 128)      { W = Ws1; Krows = 32; Ncols = 64; NT = 4; off = 0;    base = g; }
    else if (g < 192) { W = Ws2; Krows = 64; Ncols = 16; NT = 1; off = 2048; base = g - 128; }
    else if (g < 320) { W = Wc1; Krows = 31; Ncols = 64; NT = 4; off = 3072; base = g - 192; }
    else if (g < 576) { W = Wc2; Krows = 64; Ncols = 64; NT = 4; off = 5120; base = g - 320; }
    else              { W = Wc3; Krows = 64; Ncols = 3;  NT = 1; off = 9216; base = g - 576; }
    int lane = base & 31, tile = base >> 5;
    int kt = tile / NT, nt = tile % NT;
    half_t* dst = packed + off + (size_t)base * 16;
#pragma unroll
    for (int h = 0; h < 16; ++h) {
        int K = kt * 32 + ((lane < 16) ? 0 : 8) + h + ((h < 8) ? 0 : 8);
        int N = nt * 16 + (lane & 15);
        float v = (K < Krows && N < Ncols) ? W[K * Ncols + N] : 0.0f;
        dst[h] = (half_t)v;
    }
}

// ---------------- Kernel 1: double trilinear gather + SH basis ----------------

__global__ void __launch_bounds__(256) featurize(
    const float* __restrict__ rays_o, const float* __restrict__ rays_d,
    const float* __restrict__ G1, const float* __restrict__ Fgrid,
    half_t* __restrict__ featH, half_t* __restrict__ shH) {
    int p   = blockIdx.x * 256 + threadIdx.x;   // exact grid: NPTS/256 blocks
    int ray = p >> 8;
    int s   = p & 255;

    float ox = rays_o[ray * 3 + 0], oy = rays_o[ray * 3 + 1], oz = rays_o[ray * 3 + 2];
    float dx = rays_d[ray * 3 + 0], dy = rays_d[ray * 3 + 1], dz = rays_d[ray * 3 + 2];
    float inv = rsqrtf(dx * dx + dy * dy + dz * dz);
    dx *= inv; dy *= inv; dz *= inv;

    float t  = ((float)s + 0.5f) * STEPF;
    float px = ox + dx * t, py = oy + dy * t, pz = oz + dz * t;

    float gx = (px / RADIUSF + 1.0f) * (RES * 0.5f);
    float gy = (py / RADIUSF + 1.0f) * (RES * 0.5f);
    float gz = (pz / RADIUSF + 1.0f) * (RES * 0.5f);
    float tlx = floorf(gx), tly = floorf(gy), tlz = floorf(gz);
    float offx = gx - tlx, offy = gy - tly, offz = gz - tlz;
    int ix0 = (int)tlx, iy0 = (int)tly, iz0 = (int)tlz;

    v4f acc[8];
#pragma unroll
    for (int q = 0; q < 8; ++q) acc[q] = (v4f){0.f, 0.f, 0.f, 0.f};

    for (int j = 0; j < 8; ++j) {                       // 8 G1 neighbors
        int bx = (j >> 2) & 1, by = (j >> 1) & 1, bz = j & 1;
        int n0 = clampi(ix0 + bx, 0, RES - 1);
        int n1 = clampi(iy0 + by, 0, RES - 1);
        int n2 = clampi(iz0 + bz, 0, RES - 1);
        float w8 = (bx ? offx : 1.f - offx) * (by ? offy : 1.f - offy) * (bz ? offz : 1.f - offz);
        const float* gv = G1 + (size_t)((n0 * RES + n1) * RES + n2) * 3;
        float cx = gv[0], cy = gv[1], cz = gv[2];
        // grid_sample into Fgrid (coord0 -> W axis (fastest), coord2 -> D axis)
        float fix = ((cx + 1.f) * NF - 1.f) * 0.5f;
        float fiy = ((cy + 1.f) * NF - 1.f) * 0.5f;
        float fiz = ((cz + 1.f) * NF - 1.f) * 0.5f;
        float fx0 = floorf(fix), fy0 = floorf(fiy), fz0 = floorf(fiz);
        float fx = fix - fx0, fy = fiy - fy0, fz = fiz - fz0;
        int X0 = (int)fx0, Y0 = (int)fy0, Z0 = (int)fz0;
#pragma unroll
        for (int c = 0; c < 8; ++c) {                   // 8 Fgrid corners
            int cdx = c & 1, cdy = (c >> 1) & 1, cdz = (c >> 2) & 1;
            int Xi = X0 + cdx, Yi = Y0 + cdy, Zi = Z0 + cdz;
            bool valid = (Xi >= 0) & (Xi < NF) & (Yi >= 0) & (Yi < NF) & (Zi >= 0) & (Zi < NF);
            float w = w8 * (cdx ? fx : 1.f - fx) * (cdy ? fy : 1.f - fy) * (cdz ? fz : 1.f - fz)
                        * (valid ? 1.f : 0.f);
            int Xc = clampi(Xi, 0, NF - 1), Yc = clampi(Yi, 0, NF - 1), Zc = clampi(Zi, 0, NF - 1);
            const v4f* f = (const v4f*)(Fgrid + (size_t)(((Zc * NF + Yc) * NF + Xc)) * FDIM);
#pragma unroll
            for (int q = 0; q < 8; ++q) acc[q] = acc[q] + f[q] * w;
        }
    }

    // store feat as f16, row-major [point][32] (WMMA-A chunk friendly)
    const float* af = (const float*)acc;
    half_t* fd = featH + (size_t)p * 32;
#pragma unroll
    for (int q = 0; q < 4; ++q) {
        v8h hv;
#pragma unroll
        for (int i = 0; i < 8; ++i) hv[i] = (half_t)af[q * 8 + i];
        *(v8h*)(fd + q * 8) = hv;
    }

    // SH deg-4 basis (16 terms) from normalized ray dir
    float x = dx, y = dy, z = dz;
    float xx = x * x, yy = y * y, zz = z * z, xy = x * y, yz = y * z, xz = x * z;
    float sh[16];
    sh[0]  = 0.28209479177387814f;
    sh[1]  = -0.48860251190291987f * y;
    sh[2]  =  0.48860251190291987f * z;
    sh[3]  = -0.48860251190291987f * x;
    sh[4]  =  1.0925484305920792f * xy;
    sh[5]  = -1.0925484305920792f * yz;
    sh[6]  =  0.94617469575756f * zz - 0.31539156525252f;
    sh[7]  = -1.0925484305920792f * xz;
    sh[8]  =  0.5462742152960396f * (xx - yy);
    sh[9]  = -0.5900435899266435f * y * (3.0f * xx - yy);
    sh[10] =  2.890611442640554f * xy * z;
    sh[11] = -0.4570457994644657f * y * (4.0f * zz - xx - yy);
    sh[12] =  0.37317633259011546f * z * (2.0f * zz - 3.0f * xx - 3.0f * yy);
    sh[13] = -0.4570457994644657f * x * (4.0f * zz - xx - yy);
    sh[14] =  1.445305721320277f * z * (xx - yy);
    sh[15] = -0.5900435899266435f * x * (xx - 3.0f * yy);
    half_t* sd = shH + (size_t)p * 16;
#pragma unroll
    for (int q = 0; q < 2; ++q) {
        v8h hv;
#pragma unroll
        for (int i = 0; i < 8; ++i) hv[i] = (half_t)sh[q * 8 + i];
        *(v8h*)(sd + q * 8) = hv;
    }
}

// ---------------- Kernel 2: MLP chain via WMMA, one 16-point tile per wave ----------------

__global__ void __launch_bounds__(256) mlp_wmma(
    const half_t* __restrict__ featH, const half_t* __restrict__ shH,
    const half_t* __restrict__ packed,
    float* __restrict__ sigmaG, float* __restrict__ colG) {
    __shared__ __attribute__((aligned(32))) half_t act[8][2][16 * 64]; // 8 waves x ping/pong x (16 rows x 64 halves)

    int lane = threadIdx.x & 31;
    int wave = threadIdx.x >> 5;
    int tileBase = blockIdx.x * 128 + wave * 16;
    int M    = lane & 15;                 // row for A-operand fetch, column N in C/D layout
    int off8 = (lane < 16) ? 0 : 8;      // A-operand K offset / D-matrix M offset
    half_t* a0 = act[wave][0];
    half_t* a1 = act[wave][1];

    // ---- Layer 1: feat[16x32] @ Ws1[32x64] -> relu -> a0 ----
    const half_t* frow = featH + (size_t)(tileBase + M) * 32;
    v16h aL1 = load_a16(frow + off8, frow + off8 + 16);
    v8f d1[4];
#pragma unroll
    for (int nt = 0; nt < 4; ++nt) {
        v16h b = *(const v16h*)(packed + 0 + ((size_t)(nt * 32 + lane)) * 16);
        v8f c = {};
        d1[nt] = wmma16(aL1, b, c);
    }
#pragma unroll
    for (int nt = 0; nt < 4; ++nt)
#pragma unroll
        for (int r = 0; r < 8; ++r)
            a0[(r + off8) * 64 + nt * 16 + M] = (half_t)fmaxf(d1[nt][r], 0.f);
    __syncthreads();

    // ---- Layer 2: h1[16x64] @ Ws2[64x16] -> f16out[16x16] (no relu) ----
    v8f d2 = {};
#pragma unroll
    for (int kt = 0; kt < 2; ++kt) {
        const half_t* ar = a0 + M * 64 + kt * 32;
        v16h a = load_a16(ar + off8, ar + off8 + 16);
        v16h b = *(const v16h*)(packed + 2048 + ((size_t)(kt * 32 + lane)) * 16);
        d2 = wmma16(a, b, d2);
    }
    // sigma = col 0 (raw, f32); cin K=16..30 <- cols 1..15; cin K=31 <- 0
    if (M == 0) {
#pragma unroll
        for (int r = 0; r < 8; ++r) sigmaG[tileBase + r + off8] = d2[r];
    } else {
#pragma unroll
        for (int r = 0; r < 8; ++r) a1[(r + off8) * 64 + 16 + (M - 1)] = (half_t)d2[r];
    }
    if (lane < 16) a1[lane * 64 + 31] = (half_t)0.f;
    __syncthreads();

    // ---- Layer 3: cin[16x32] (K0..15 = SH from global, K16..31 from LDS) @ Wc1 -> relu -> a0 ----
    const half_t* shrow = shH + (size_t)(tileBase + M) * 16;
    v16h aL3 = load_a16(shrow + off8, a1 + M * 64 + 16 + off8);
    v8f d3[4];
#pragma unroll
    for (int nt = 0; nt < 4; ++nt) {
        v16h b = *(const v16h*)(packed + 3072 + ((size_t)(nt * 32 + lane)) * 16);
        v8f c = {};
        d3[nt] = wmma16(aL3, b, c);
    }
    __syncthreads();   // a0 reads of layer 2 done before overwrite
#pragma unroll
    for (int nt = 0; nt < 4; ++nt)
#pragma unroll
        for (int r = 0; r < 8; ++r)
            a0[(r + off8) * 64 + nt * 16 + M] = (half_t)fmaxf(d3[nt][r], 0.f);
    __syncthreads();

    // ---- Layer 4: h3[16x64] @ Wc2[64x64] -> relu -> a1 ----
    v8f d4[4];
#pragma unroll
    for (int nt = 0; nt < 4; ++nt) d4[nt] = (v8f){};
#pragma unroll
    for (int kt = 0; kt < 2; ++kt) {
        const half_t* ar = a0 + M * 64 + kt * 32;
        v16h a = load_a16(ar + off8, ar + off8 + 16);
#pragma unroll
        for (int nt = 0; nt < 4; ++nt) {
            v16h b = *(const v16h*)(packed + 5120 + ((size_t)((kt * 4 + nt) * 32 + lane)) * 16);
            d4[nt] = wmma16(a, b, d4[nt]);
        }
    }
    __syncthreads();   // a1 reads of layer 3 done before overwrite
#pragma unroll
    for (int nt = 0; nt < 4; ++nt)
#pragma unroll
        for (int r = 0; r < 8; ++r)
            a1[(r + off8) * 64 + nt * 16 + M] = (half_t)fmaxf(d4[nt][r], 0.f);
    __syncthreads();

    // ---- Layer 5: h4[16x64] @ Wc3[64x3->16] -> craw (cols 0..2) ----
    v8f d5 = {};
#pragma unroll
    for (int kt = 0; kt < 2; ++kt) {
        const half_t* ar = a1 + M * 64 + kt * 32;
        v16h a = load_a16(ar + off8, ar + off8 + 16);
        v16h b = *(const v16h*)(packed + 9216 + ((size_t)(kt * 32 + lane)) * 16);
        d5 = wmma16(a, b, d5);
    }
    if (M < 3) {
#pragma unroll
        for (int r = 0; r < 8; ++r)
            colG[(size_t)(tileBase + r + off8) * 3 + M] = d5[r];
    }
}

// ---------------- Kernel 3: per-ray compositing ----------------

__global__ void __launch_bounds__(256) composite(
    const float* __restrict__ rays_o, const float* __restrict__ rays_d,
    const float* __restrict__ sigmaG, const float* __restrict__ colG,
    float* __restrict__ out) {
    int ray = blockIdx.x * blockDim.x + threadIdx.x;
    if (ray >= NRAYS) return;
    float ox = rays_o[ray * 3 + 0], oy = rays_o[ray * 3 + 1], oz = rays_o[ray * 3 + 2];
    float dx = rays_d[ray * 3 + 0], dy = rays_d[ray * 3 + 1], dz = rays_d[ray * 3 + 2];
    float inv = rsqrtf(dx * dx + dy * dy + dz * dz);
    dx *= inv; dy *= inv; dz *= inv;

    float T = 1.f, wsum = 0.f, ar = 0.f, ag = 0.f, ab = 0.f;
    for (int s = 0; s < NI; ++s) {
        float t  = ((float)s + 0.5f) * STEPF;
        float px = ox + dx * t, py = oy + dy * t, pz = oz + dz * t;
        bool mask = fmaxf(fabsf(px), fmaxf(fabsf(py), fabsf(pz))) <= RADIUSF;
        float sg = sigmaG[ray * NI + s];
        sg = mask ? fmaxf(sg, 0.f) : 0.f;
        float alpha = 1.f - __expf(-sg * STEPF);
        float w = alpha * T;
        const float* c = colG + (size_t)(ray * NI + s) * 3;
        float cr = mask ? c[0] : 0.f;
        float cg = mask ? c[1] : 0.f;
        float cb = mask ? c[2] : 0.f;
        ar += w / (1.f + __expf(-cr));
        ag += w / (1.f + __expf(-cg));
        ab += w / (1.f + __expf(-cb));
        T *= (1.f - alpha + 1e-10f);
        wsum += w;
    }
    float bg = 1.f - wsum;
    out[ray * 3 + 0] = ar + bg;
    out[ray * 3 + 1] = ag + bg;
    out[ray * 3 + 2] = ab + bg;
}

// ---------------- Launch ----------------

extern "C" void kernel_launch(void* const* d_in, const int* in_sizes, int n_in,
                              void* d_out, int out_size, void* d_ws, size_t ws_size,
                              hipStream_t stream) {
    const float* rays_o = (const float*)d_in[0];
    const float* rays_d = (const float*)d_in[1];
    const float* G1     = (const float*)d_in[2];
    const float* Fgrid  = (const float*)d_in[3];
    const float* Ws1    = (const float*)d_in[4];
    const float* Ws2    = (const float*)d_in[5];
    const float* Wc1    = (const float*)d_in[6];
    const float* Wc2    = (const float*)d_in[7];
    const float* Wc3    = (const float*)d_in[8];
    float* out = (float*)d_out;

    char* ws = (char*)d_ws;
    size_t o = 0;
    half_t* packed = (half_t*)(ws + o); o += 32768;                       // 20KB used
    half_t* featH  = (half_t*)(ws + o); o += (size_t)NPTS * 32 * 2;       // 16MB
    half_t* shH    = (half_t*)(ws + o); o += (size_t)NPTS * 16 * 2;       // 8MB
    float*  sigmaG = (float*) (ws + o); o += (size_t)NPTS * 4;            // 1MB
    float*  colG   = (float*) (ws + o); o += (size_t)NPTS * 12;           // 3MB

    prepack_weights<<<3, 256, 0, stream>>>(Ws1, Ws2, Wc1, Wc2, Wc3, packed);
    featurize<<<NPTS / 256, 256, 0, stream>>>(rays_o, rays_d, G1, Fgrid, featH, shH);
    mlp_wmma<<<NPTS / 128, 256, 0, stream>>>(featH, shH, packed, sigmaG, colG);
    composite<<<NRAYS / 256, 256, 0, stream>>>(rays_o, rays_d, sigmaG, colG, out);
}
